// CausalSelfAttention_87935160418946
// MI455X (gfx1250) — matrix operand order repaired and checked by reading, not compiled, not run on hardware
//
#include <hip/hip_runtime.h>
#include <hip/hip_bf16.h>
#include <stdint.h>

// ---------------------------------------------------------------------------
// Causal self-attention forward for MI455X (gfx1250, wave32, WMMA bf16).
// B=4, T=2048, C=1024, H=16, HS=64.
// Pipeline: cvt/transpose(f32->bf16) -> QKV GEMM (WMMA+TDM) ->
//           flash attention (WMMA+TDM) -> projection GEMM (WMMA+TDM, fp32 out).
// All matrix operands are staged so that every LDS fragment read is
// row-contiguous (ds_load_b128): weights are pre-transposed, V is stored
// transposed per head. Tile staging uses the Tensor Data Mover
// (tensor_load_to_lds + s_wait_tensorcnt), double-buffered.
// ---------------------------------------------------------------------------

typedef __attribute__((ext_vector_type(16))) __bf16 v16bf;
typedef __attribute__((ext_vector_type(8)))  float  v8f;
typedef __attribute__((ext_vector_type(4)))  unsigned int u32x4;
typedef __attribute__((ext_vector_type(8)))  int i32x8;
typedef __attribute__((ext_vector_type(4)))  int i32x4;

static constexpr int Bsz   = 4;
static constexpr int T     = 2048;
static constexpr int C     = 1024;
static constexpr int H     = 16;
static constexpr int HS    = 64;
static constexpr int MROWS = Bsz * T;            // 8192
static constexpr int QKVN  = Bsz * H * T * HS;   // elements per q/k/v matrix

#define WMMA_BF16(a, b, c) \
  __builtin_amdgcn_wmma_f32_16x16x32_bf16(false, (a), false, (b), (short)0, (c), false, false)

#if defined(__has_builtin)
#  if __has_builtin(__builtin_amdgcn_tensor_load_to_lds)
#    define HAVE_TDM 1
#  else
#    define HAVE_TDM 0
#  endif
#else
#  define HAVE_TDM 0
#endif

__device__ __forceinline__ v8f zero8() {
  v8f z = {0.f, 0.f, 0.f, 0.f, 0.f, 0.f, 0.f, 0.f};
  return z;
}

#if HAVE_TDM
// Low 32 bits of a generic LDS pointer are the LDS byte offset
// (LDS aperture is {SHARED_BASE,16'b0} in addr[63:32]).
__device__ __forceinline__ unsigned int lds_off(const void* p) {
  return (unsigned int)(uintptr_t)p;
}

// 2D strided tile (bf16 elements) -> contiguous LDS, via TDM (D# per ISA 8.3/8.4).
// tensor dims are set equal to the tile (tiles never cross bounds -> no OOB fill).
__device__ __forceinline__ void tdm_load_2d(unsigned int lds_byte_off,
                                            const void* gptr,
                                            unsigned int tile_w,      // elems
                                            unsigned int tile_h,      // rows
                                            unsigned int row_stride)  // elems
{
  const unsigned long long ga = (unsigned long long)(uintptr_t)gptr;
  u32x4 g0;
  g0[0] = 1u;                                    // count=1 (valid), user mode
  g0[1] = lds_byte_off;                          // lds_addr
  g0[2] = (unsigned int)ga;                      // global_addr[31:0]
  g0[3] = ((unsigned int)(ga >> 32) & 0x01FFFFFFu) | (2u << 30);  // type=2
  i32x8 g1;
  g1[0] = (int)(1u << 16);                       // data_size=1 (2 bytes)
  g1[1] = (int)((tile_w & 0xFFFFu) << 16);                       // tensor_dim0 lo
  g1[2] = (int)(((tile_w >> 16) & 0xFFFFu) | ((tile_h & 0xFFFFu) << 16)); // d0 hi | d1 lo
  g1[3] = (int)(((tile_h >> 16) & 0xFFFFu) | ((tile_w & 0xFFFFu) << 16)); // d1 hi | tile_dim0
  g1[4] = (int)(tile_h & 0xFFFFu);               // tile_dim1 (tile_dim2 = 0)
  g1[5] = (int)row_stride;                       // tensor_dim0_stride[31:0]
  g1[6] = 0;                                     // stride hi | dim1_stride lo
  g1[7] = 0;                                     // dim1_stride hi
  const i32x4 z4 = {0, 0, 0, 0};
#if __clang_major__ >= 23
  const i32x8 z8 = {0, 0, 0, 0, 0, 0, 0, 0};
  __builtin_amdgcn_tensor_load_to_lds(g0, g1, z4, z4, z8, 0);
#else
  __builtin_amdgcn_tensor_load_to_lds(g0, g1, z4, z4, 0);
#endif
}
#endif  // HAVE_TDM

// ---- fragment loaders (wave32), contiguous k order: k = 16*(lane/16) + element -----------
// Both loaders read 16 row-contiguous bf16 per lane -> ds_load_b128 x2.

// A-matrix 16x32 bf16: lane(0-15)=row M, lanes 16-31 same rows with K 16-31.
__device__ __forceinline__ v16bf frag_a(const __bf16* p, int ld) {
  const int lane = threadIdx.x & 31;
  const int m    = lane & 15;
  const int kb   = (lane >> 4) << 4;          // 0 or 16
  const __bf16* row = p + m * ld;
  v16bf a;
#pragma unroll
  for (int j = 0; j < 8; ++j) {
    const int kk = kb + 2 * j;
    a[2 * j]     = row[kk];
    a[2 * j + 1] = row[kk + 1];
  }
  return a;
}

// B fragment from a row-major [16 x >=32] tile X holding B^T (B[kk][n]=X[n][kk]):
// lane n reads row n contiguously; lanes 16-31 take K 16-31.
__device__ __forceinline__ v16bf frag_bT(const __bf16* p, int ld) {
  const int lane = threadIdx.x & 31;
  const int n    = lane & 15;
  const int kb   = lane & 16;
  const __bf16* row = p + n * ld;
  v16bf b;
#pragma unroll
  for (int j = 0; j < 8; ++j) {
    const int kk = kb + 2 * j;
    b[2 * j]     = row[kk];
    b[2 * j + 1] = row[kk + 1];
  }
  return b;
}

// ---- fp32 -> bf16 conversion (plain + transposing) -------------------------

__global__ void cvt_kernel(const float* __restrict__ in, __bf16* __restrict__ out, int n) {
  int i = blockIdx.x * blockDim.x + threadIdx.x;
  const int stride = gridDim.x * blockDim.x;
  for (; i < n; i += stride) out[i] = (__bf16)in[i];
}

// in: [R][Ccols] fp32 -> out: [Ccols][R] bf16 (32x32 LDS tiles, 32x8 threads)
__global__ void __launch_bounds__(256)
cvtT_kernel(const float* __restrict__ in, __bf16* __restrict__ out, int R, int Ccols) {
  __shared__ float tile[32][33];
  const int c0 = blockIdx.x * 32;
  const int r0 = blockIdx.y * 32;
  const int tx = threadIdx.x, ty = threadIdx.y;
#pragma unroll
  for (int u = 0; u < 4; ++u)
    tile[ty + u * 8][tx] = in[(size_t)(r0 + ty + u * 8) * Ccols + c0 + tx];
  __syncthreads();
#pragma unroll
  for (int u = 0; u < 4; ++u)
    out[(size_t)(c0 + ty + u * 8) * R + r0 + tx] = (__bf16)tile[tx][ty + u * 8];
}

// ---- tiled WMMA GEMM: [M,K] x W^T[N,K] + bias; tile 128x128, BK=32 ---------
// MODE 0: scatter bf16 into qkv (q,k: [B][H][T][HS]; v transposed [B][H][HS][T]).
// MODE 1: fp32 row-major out.

template <int MODE>
__global__ void __launch_bounds__(256)
gemm_bf16(const __bf16* __restrict__ A, const __bf16* __restrict__ Wt,
          const float* __restrict__ bias, void* __restrict__ Out,
          int Ndim, int Kdim) {
  __shared__ __bf16 sA[2][128 * 32];
  __shared__ __bf16 sB[2][128 * 32];  // W^T tile: [128 n-rows][32 k]
  const int tid  = threadIdx.x;
  const int wid  = tid >> 5;
  const int lane = tid & 31;
  const int rowBase = blockIdx.x * 128;
  const int colBase = blockIdx.y * 128;

  v8f acc[8];
#pragma unroll
  for (int i = 0; i < 8; ++i) acc[i] = zero8();

  int cur = 0;
#if HAVE_TDM
  if (tid < 32) {  // wave 0 drives the TDM pipeline
    tdm_load_2d(lds_off(&sA[0][0]), A + (size_t)rowBase * Kdim, 32, 128, Kdim);
    tdm_load_2d(lds_off(&sB[0][0]), Wt + (size_t)colBase * Kdim, 32, 128, Kdim);
  }
#endif

  for (int k0 = 0; k0 < Kdim; k0 += 32) {
#if HAVE_TDM
    if (tid < 32) {
      if (k0 + 32 < Kdim) {  // prefetch next tiles into the other buffer
        tdm_load_2d(lds_off(&sA[cur ^ 1][0]),
                    A + (size_t)rowBase * Kdim + (k0 + 32), 32, 128, Kdim);
        tdm_load_2d(lds_off(&sB[cur ^ 1][0]),
                    Wt + (size_t)colBase * Kdim + (k0 + 32), 32, 128, Kdim);
        __builtin_amdgcn_s_wait_tensorcnt(2);  // current pair done (in-order)
      } else {
        __builtin_amdgcn_s_wait_tensorcnt(0);
      }
    }
    __syncthreads();
#else
    {  // A tile 128x32
      const int r = tid >> 1, c = (tid & 1) << 4;
      const uint4* src = (const uint4*)(A + (size_t)(rowBase + r) * Kdim + k0 + c);
      uint4* dst = (uint4*)(&sA[cur][r * 32 + c]);
      dst[0] = src[0];
      dst[1] = src[1];
    }
    {  // W^T tile 128x32
      const int r = tid >> 1, c = (tid & 1) << 4;
      const uint4* src = (const uint4*)(Wt + (size_t)(colBase + r) * Kdim + k0 + c);
      uint4* dst = (uint4*)(&sB[cur][r * 32 + c]);
      dst[0] = src[0];
      dst[1] = src[1];
    }
    __syncthreads();
#endif

    const v16bf a = frag_a(&sA[cur][wid * 16 * 32], 32);
#pragma unroll
    for (int nt = 0; nt < 8; ++nt)
      acc[nt] = WMMA_BF16(a, frag_bT(&sB[cur][nt * 16 * 32], 32), acc[nt]);

    __syncthreads();  // bounds wave skew; frees buffer cur
    cur ^= 1;
  }

  const int mhi = (lane >> 4) << 3;
#pragma unroll
  for (int nt = 0; nt < 8; ++nt) {
#pragma unroll
    for (int i = 0; i < 8; ++i) {
      const int r = rowBase + wid * 16 + mhi + i;
      const int j = colBase + nt * 16 + (lane & 15);
      const float v = acc[nt][i] + bias[j];
      if (MODE == 0) {
        __bf16* qkv = (__bf16*)Out;
        const int which = j >> 10, rem = j & 1023;
        const int h = rem >> 6, d = rem & 63;
        const int bb = r >> 11, tt = r & 2047;
        if (which == 2)  // V stored transposed: [B][H][HS][T]
          qkv[(size_t)2 * QKVN + (((bb * H + h) * HS + d) * T + tt)] = (__bf16)v;
        else
          qkv[(size_t)which * QKVN + (((bb * H + h) * T + tt) * HS + d)] = (__bf16)v;
      } else {
        ((float*)Out)[(size_t)r * Ndim + j] = v;
      }
    }
  }
}

// ---- flash attention: block = (b, h, 64-query tile), 4 waves x 16 rows -----

__global__ void __launch_bounds__(128)
attn_kernel(const __bf16* __restrict__ qkv, __bf16* __restrict__ y) {
  __shared__ __bf16 sK[2][64 * 64];   // [key][hs]
  __shared__ __bf16 sVt[2][64 * 64];  // [hs][key]  (V stored transposed)
  __shared__ __bf16 sP[64 * 64];      // per-wave 16x64 P scratch

  const int tid  = threadIdx.x;
  const int wid  = tid >> 5;
  const int lane = tid & 31;
  const int qblk = blockIdx.x & 31;  // T/64 = 32 query tiles
  const int h    = (blockIdx.x >> 5) & 15;
  const int b    = blockIdx.x >> 9;

  const size_t headBase = (size_t)(b * H + h) * T * HS;
  const __bf16* Q  = qkv + headBase;
  const __bf16* Kp = qkv + (size_t)QKVN + headBase;       // [T][HS]
  const __bf16* Vt = qkv + (size_t)2 * QKVN + headBase;   // [HS][T]

  const __bf16* qrow = Q + (size_t)(qblk * 64 + wid * 16) * HS;
  const v16bf qa0 = frag_a(qrow, HS);
  const v16bf qa1 = frag_a(qrow + 32, HS);

  float rm[8], rs[8];
  v8f of[4];
#pragma unroll
  for (int i = 0; i < 8; ++i) { rm[i] = -1e30f; rs[i] = 0.f; }
#pragma unroll
  for (int i = 0; i < 4; ++i) of[i] = zero8();

  const int mhi = (lane >> 4) << 3;
  const float scale = 0.125f;  // 1/sqrt(64)

  int cur = 0;
#if HAVE_TDM
  if (tid < 32) {
    tdm_load_2d(lds_off(&sK[0][0]), Kp, 4096, 1, 4096);          // contiguous
    tdm_load_2d(lds_off(&sVt[0][0]), Vt, 64, 64, T);             // 64 rows, stride T
  }
#endif

  for (int kb = 0; kb <= qblk; ++kb) {
#if HAVE_TDM
    if (tid < 32) {
      if (kb + 1 <= qblk) {
        tdm_load_2d(lds_off(&sK[cur ^ 1][0]), Kp + (size_t)(kb + 1) * 4096, 4096, 1, 4096);
        tdm_load_2d(lds_off(&sVt[cur ^ 1][0]), Vt + (size_t)(kb + 1) * 64, 64, 64, T);
        __builtin_amdgcn_s_wait_tensorcnt(2);
      } else {
        __builtin_amdgcn_s_wait_tensorcnt(0);
      }
    }
    __syncthreads();
#else
    {
      const int r = tid >> 1, c = (tid & 1) << 5;
      const uint4* ks = (const uint4*)(Kp + (size_t)(kb * 64 + r) * HS + c);
      const uint4* vs = (const uint4*)(Vt + (size_t)r * T + kb * 64 + c);
      uint4* kd = (uint4*)(&sK[cur][r * 64 + c]);
      uint4* vd = (uint4*)(&sVt[cur][r * 64 + c]);
#pragma unroll
      for (int u = 0; u < 4; ++u) { kd[u] = ks[u]; vd[u] = vs[u]; }
    }
    __syncthreads();
#endif

    // S = Q * K^T (16x64 per wave: 4 N-tiles x 2 K-chunks)
    v8f s[4];
#pragma unroll
    for (int nt = 0; nt < 4; ++nt) {
      v8f z = zero8();
      z = WMMA_BF16(qa0, frag_bT(&sK[cur][nt * 16 * 64], 64), z);
      z = WMMA_BF16(qa1, frag_bT(&sK[cur][nt * 16 * 64 + 32], 64), z);
      s[nt] = z;
    }

    // scale + causal mask (only the diagonal block)
#pragma unroll
    for (int nt = 0; nt < 4; ++nt) {
#pragma unroll
      for (int i = 0; i < 8; ++i) {
        float v = s[nt][i] * scale;
        if (kb == qblk) {
          const int kidx = nt * 16 + (lane & 15);
          const int qidx = wid * 16 + mhi + i;
          if (kidx > qidx) v = -1e30f;
        }
        s[nt][i] = v;
      }
    }

    // online softmax (rows live across 16-lane halves of the C fragment)
#pragma unroll
    for (int i = 0; i < 8; ++i) {
      float t = fmaxf(fmaxf(s[0][i], s[1][i]), fmaxf(s[2][i], s[3][i]));
#pragma unroll
      for (int mm = 1; mm < 16; mm <<= 1) t = fmaxf(t, __shfl_xor(t, mm, 32));
      const float mnew  = fmaxf(rm[i], t);
      const float alpha = __expf(rm[i] - mnew);
      rm[i] = mnew;
      float ps = 0.f;
#pragma unroll
      for (int nt = 0; nt < 4; ++nt) {
        const float p = __expf(s[nt][i] - mnew);
        s[nt][i] = p;
        ps += p;
      }
#pragma unroll
      for (int mm = 1; mm < 16; mm <<= 1) ps += __shfl_xor(ps, mm, 32);
      rs[i] = rs[i] * alpha + ps;
#pragma unroll
      for (int ht = 0; ht < 4; ++ht) of[ht][i] *= alpha;
    }

    // P (C-frag) -> LDS -> A-frag; O += P * V  (V^T rows are contiguous)
    __bf16* P = sP + wid * 16 * 64;
#pragma unroll
    for (int nt = 0; nt < 4; ++nt)
#pragma unroll
      for (int i = 0; i < 8; ++i)
        P[(mhi + i) * 64 + nt * 16 + (lane & 15)] = (__bf16)s[nt][i];

    const v16bf pa0 = frag_a(P, 64);
    const v16bf pa1 = frag_a(P + 32, 64);
#pragma unroll
    for (int ht = 0; ht < 4; ++ht) {
      of[ht] = WMMA_BF16(pa0, frag_bT(&sVt[cur][ht * 16 * 64], 64), of[ht]);
      of[ht] = WMMA_BF16(pa1, frag_bT(&sVt[cur][ht * 16 * 64 + 32], 64), of[ht]);
    }

    __syncthreads();
    cur ^= 1;
  }

  // normalize and write y (bf16, [B,T,C] with this head's HS columns)
#pragma unroll
  for (int i = 0; i < 8; ++i) {
    const float inv = 1.0f / rs[i];
    const int q = qblk * 64 + wid * 16 + mhi + i;
#pragma unroll
    for (int ht = 0; ht < 4; ++ht) {
      const int d = ht * 16 + (lane & 15);
      y[((size_t)(b * T + q)) * C + h * HS + d] = (__bf16)(of[ht][i] * inv);
    }
  }
}

// ---------------------------------------------------------------------------

extern "C" void kernel_launch(void* const* d_in, const int* in_sizes, int n_in,
                              void* d_out, int out_size, void* d_ws, size_t ws_size,
                              hipStream_t stream) {
  const float* x      = (const float*)d_in[0];
  const float* w_attn = (const float*)d_in[1];
  const float* b_attn = (const float*)d_in[2];
  const float* w_proj = (const float*)d_in[3];
  const float* b_proj = (const float*)d_in[4];
  float* out = (float*)d_out;

  // workspace layout (bf16 elements), total ~88 MiB
  __bf16* xb  = (__bf16*)d_ws;                 // [8192,1024]
  __bf16* wab = xb  + (size_t)MROWS * C;       // W_attn^T [3072,1024]
  __bf16* wpb = wab + (size_t)C * 3 * C;       // W_proj^T [1024,1024]
  __bf16* qkv = wpb + (size_t)C * C;           // q,k:[B][H][T][HS]; v:[B][H][HS][T]
  __bf16* yb  = qkv + (size_t)3 * QKVN;        // [8192,1024]

  cvt_kernel<<<(MROWS * C + 1023) / 1024, 256, 0, stream>>>(x, xb, MROWS * C);
  cvtT_kernel<<<dim3(3 * C / 32, C / 32), dim3(32, 8), 0, stream>>>(w_attn, wab, C, 3 * C);
  cvtT_kernel<<<dim3(C / 32, C / 32), dim3(32, 8), 0, stream>>>(w_proj, wpb, C, C);

  gemm_bf16<0><<<dim3(MROWS / 128, (3 * C) / 128), 256, 0, stream>>>(
      xb, wab, b_attn, (void*)qkv, 3 * C, C);

  attn_kernel<<<Bsz * H * (T / 64), 128, 0, stream>>>(qkv, yb);

  gemm_bf16<1><<<dim3(MROWS / 128, C / 128), 256, 0, stream>>>(
      yb, wpb, b_proj, (void*)out, C, C);
}
